// IG_MSA_M_43894565765143
// MI455X (gfx1250) — matrix operand order repaired and checked by reading, not compiled, not run on hardware
//
#include <hip/hip_runtime.h>
#include <hip/hip_bf16.h>
#include <math.h>

// IG-MSA (Retinexformer illumination-guided channel attention), MI455X/gfx1250.
// All GEMM stages run on v_wmma_f32_16x16x32_bf16 (bf16 in, f32 accumulate).
// Pipeline is HBM-bandwidth-bound (~1.2 GB moved @ 23.3 TB/s) once matmuls
// are on the WMMA pipe; intermediates kept bf16 in workspace to halve traffic.

#define BATCH 8
#define NTOK 16384
#define CH 256
#define HEADS 4
#define DHEAD 64

typedef __attribute__((ext_vector_type(16))) __bf16 v16bf;
typedef __attribute__((ext_vector_type(8)))  float  v8f;

// A-fragment K index for 16-bit 16x32 A tiles (ISA 7.12.2):
// lanes 0-15: V0..3 -> K0..7, V4..7 -> K16..23 ; lanes 16-31: +8
__device__ __forceinline__ int a_kidx(int lane, int j) {
    return ((j < 8) ? j : (j + 8)) + ((lane >= 16) ? 8 : 0);
}

__global__ void zero_f32(float* p, int n) {
    int i = blockIdx.x * blockDim.x + threadIdx.x;
    if (i < n) p[i] = 0.f;
}

// ---------------- Stage A: Q,K,V = X @ {Wq,Wk,Wv}  (bf16 WMMA) --------------
__global__ __launch_bounds__(256) void qkv_gemm(
    const float* __restrict__ X, const float* __restrict__ Wq,
    const float* __restrict__ Wk, const float* __restrict__ Wv,
    __bf16* __restrict__ q_ws, __bf16* __restrict__ k_ws, __bf16* __restrict__ v_ws)
{
    __shared__ __bf16 ldsX[128 * 32];
    __shared__ __bf16 ldsW[32 * 64];
    int b  = blockIdx.z;
    int n0 = blockIdx.x * 128;
    int c0 = blockIdx.y * 64;            // 0..767 across [Wq|Wk|Wv]
    int wsel = c0 >> 8;
    int wc0  = c0 & 255;
    const float* W = (wsel == 0) ? Wq : ((wsel == 1) ? Wk : Wv);
    __bf16* OUT    = (wsel == 0) ? q_ws : ((wsel == 1) ? k_ws : v_ws);

    int tid = threadIdx.x, lane = tid & 31, wave = tid >> 5;

    v8f acc[4];
    #pragma unroll
    for (int t = 0; t < 4; ++t)
        #pragma unroll
        for (int r = 0; r < 8; ++r) acc[t][r] = 0.f;

    for (int k0 = 0; k0 < CH; k0 += 32) {
        for (int i = tid; i < 128 * 32; i += 256) {
            int r = i >> 5, kk = i & 31;
            ldsX[i] = (__bf16)X[((size_t)b * NTOK + n0 + r) * CH + k0 + kk];
        }
        for (int i = tid; i < 32 * 64; i += 256) {
            int r = i >> 6, cc = i & 63;
            ldsW[i] = (__bf16)W[(k0 + r) * CH + wc0 + cc];
        }
        __syncthreads();

        v16bf a;
        int m = lane & 15;
        #pragma unroll
        for (int j = 0; j < 16; ++j)
            a[j] = ldsX[(wave * 16 + m) * 32 + a_kidx(lane, j)];

        #pragma unroll
        for (int t = 0; t < 4; ++t) {
            v16bf bb;
            #pragma unroll
            for (int j = 0; j < 16; ++j) bb[j] = ldsW[a_kidx(lane, j) * 64 + t * 16 + (lane & 15)];
            acc[t] = __builtin_amdgcn_wmma_f32_16x16x32_bf16(
                false, a, false, bb, (short)0, acc[t], false, false);
        }
        __syncthreads();
    }

    int mbase = n0 + wave * 16 + ((lane >= 16) ? 8 : 0);
    int nb = lane & 15;
    #pragma unroll
    for (int t = 0; t < 4; ++t)
        #pragma unroll
        for (int r = 0; r < 8; ++r)
            OUT[((size_t)b * NTOK + mbase + r) * CH + wc0 + t * 16 + nb] =
                (__bf16)acc[t][r];
}

// ---------------- Stage B: per-(b,channel) sum of squares over tokens -------
__global__ __launch_bounds__(256) void col_sumsq(
    const __bf16* __restrict__ q_ws, const __bf16* __restrict__ k_ws,
    float* __restrict__ ssq, float* __restrict__ ssk)
{
    int b = blockIdx.y;
    int n0 = blockIdx.x * 64;
    int c = threadIdx.x;
    float sq = 0.f, sk = 0.f;
    for (int t = 0; t < 64; ++t) {
        size_t idx = ((size_t)b * NTOK + n0 + t) * CH + c;
        float qv = (float)q_ws[idx];
        float kv = (float)k_ws[idx];
        sq += qv * qv; sk += kv * kv;
    }
    atomicAdd(&ssq[b * CH + c], sq);
    atomicAdd(&ssk[b * CH + c], sk);
}

__global__ void make_invnorm(float* s, int n) {
    int i = blockIdx.x * blockDim.x + threadIdx.x;
    if (i < n) { float nv = sqrtf(s[i]); s[i] = 1.f / fmaxf(nv, 1e-12f); }
}

// ---------------- Stage C: S = K^T Q per (b,h), scaled by inv norms ---------
// K tile is staged into LDS transposed (d-major) so the WMMA A-fragment
// gather is two contiguous 16B runs per lane (vectorizes to ds_load_b128).
__global__ __launch_bounds__(256) void ktq_gemm(
    const __bf16* __restrict__ q_ws, const __bf16* __restrict__ k_ws,
    const float* __restrict__ invq, const float* __restrict__ invk,
    const float* __restrict__ rescale, float* __restrict__ attn)
{
    __shared__ __bf16 ldsKt[64 * 32];    // [d][n] transposed
    __shared__ __bf16 ldsQ[32 * 64];     // [n][e]
    int bh = blockIdx.x;                 // 32 total
    int b = bh >> 2, h = bh & 3;
    int tid = threadIdx.x, lane = tid & 31, wave = tid >> 5;
    int ch0 = h * DHEAD;

    v8f acc[2];
    #pragma unroll
    for (int t = 0; t < 2; ++t)
        #pragma unroll
        for (int r = 0; r < 8; ++r) acc[t][r] = 0.f;

    // tiles 2w and 2w+1 always share the same d-tile
    int d0 = ((wave * 2) >> 2) * 16;

    for (int n0 = 0; n0 < NTOK; n0 += 32) {
        for (int i = tid; i < 32 * 64; i += 256) {
            int r = i >> 6, cc = i & 63;     // r = n within chunk, cc = channel
            size_t g = ((size_t)b * NTOK + n0 + r) * CH + ch0 + cc;
            ldsKt[cc * 32 + r] = k_ws[g];    // transpose on store
            ldsQ[i] = q_ws[g];
        }
        __syncthreads();

        v16bf a;                             // A[M=d][K=n] = K[n][d]
        int m = lane & 15;
        #pragma unroll
        for (int j = 0; j < 16; ++j)
            a[j] = ldsKt[(d0 + m) * 32 + a_kidx(lane, j)];

        #pragma unroll
        for (int t = 0; t < 2; ++t) {
            int e0 = ((wave * 2 + t) & 3) * 16;
            v16bf bb;                        // B[K=n][N=e] = Q[n][e]
            #pragma unroll
            for (int j = 0; j < 16; ++j) bb[j] = ldsQ[a_kidx(lane, j) * 64 + e0 + (lane & 15)];
            acc[t] = __builtin_amdgcn_wmma_f32_16x16x32_bf16(
                false, a, false, bb, (short)0, acc[t], false, false);
        }
        __syncthreads();
    }

    float rs = rescale[h];
    #pragma unroll
    for (int t = 0; t < 2; ++t) {
        int e0 = ((wave * 2 + t) & 3) * 16;
        int e = e0 + (lane & 15);
        #pragma unroll
        for (int r = 0; r < 8; ++r) {
            int d = d0 + r + ((lane >= 16) ? 8 : 0);
            float v = acc[t][r] * invk[b * CH + ch0 + d]
                                * invq[b * CH + ch0 + e] * rs;
            attn[((size_t)bh * DHEAD + d) * DHEAD + e] = v;
        }
    }
}

// ---------------- softmax over last dim (64) --------------------------------
__global__ __launch_bounds__(64) void attn_softmax(float* attn) {
    __shared__ float buf[64];
    int row = blockIdx.x;                // bh*64 + d
    int e = threadIdx.x;
    float v = attn[(size_t)row * 64 + e];
    buf[e] = v;
    __syncthreads();
    for (int s = 32; s > 0; s >>= 1) {
        if (e < s) buf[e] = fmaxf(buf[e], buf[e + s]);
        __syncthreads();
    }
    float mx = buf[0];
    __syncthreads();
    float ex = expf(v - mx);
    buf[e] = ex;
    __syncthreads();
    for (int s = 32; s > 0; s >>= 1) {
        if (e < s) buf[e] += buf[e + s];
        __syncthreads();
    }
    float sum = buf[0];
    attn[(size_t)row * 64 + e] = ex / sum;
}

// ---------------- Stage D: X_mid[n, h*64+d] = sum_e (V*illu)[n,e] attn[d,e] -
__global__ __launch_bounds__(256) void av_gemm(
    const __bf16* __restrict__ v_ws, const float* __restrict__ illu,
    const float* __restrict__ attn, __bf16* __restrict__ xmid)
{
    __shared__ __bf16 ldsV[128 * 64];
    __shared__ __bf16 ldsAT[64 * 64];    // attn transposed: [e][d]
    int b = blockIdx.z, h = blockIdx.y;
    int n0 = blockIdx.x * 128;
    int bh = b * HEADS + h;
    int ch0 = h * DHEAD;
    int tid = threadIdx.x, lane = tid & 31, wave = tid >> 5;

    for (int i = tid; i < 128 * 64; i += 256) {
        int r = i >> 6, cc = i & 63;
        size_t g = ((size_t)b * NTOK + n0 + r) * CH + ch0 + cc;
        ldsV[i] = (__bf16)((float)v_ws[g] * illu[g]);
    }
    for (int i = tid; i < 64 * 64; i += 256) {
        int d = i >> 6, e = i & 63;
        ldsAT[e * 64 + d] = (__bf16)attn[((size_t)bh * DHEAD + d) * DHEAD + e];
    }
    __syncthreads();

    v8f acc[4];
    #pragma unroll
    for (int t = 0; t < 4; ++t)
        #pragma unroll
        for (int r = 0; r < 8; ++r) acc[t][r] = 0.f;

    #pragma unroll
    for (int k0 = 0; k0 < DHEAD; k0 += 32) {
        v16bf a;
        int m = lane & 15;
        #pragma unroll
        for (int j = 0; j < 16; ++j)
            a[j] = ldsV[(wave * 16 + m) * 64 + k0 + a_kidx(lane, j)];
        #pragma unroll
        for (int t = 0; t < 4; ++t) {
            v16bf bb;
            #pragma unroll
            for (int j = 0; j < 16; ++j)
                bb[j] = ldsAT[(k0 + a_kidx(lane, j)) * 64 + t * 16 + (lane & 15)];
            acc[t] = __builtin_amdgcn_wmma_f32_16x16x32_bf16(
                false, a, false, bb, (short)0, acc[t], false, false);
        }
    }

    int mbase = n0 + wave * 16 + ((lane >= 16) ? 8 : 0);
    int nb = lane & 15;
    #pragma unroll
    for (int t = 0; t < 4; ++t)
        #pragma unroll
        for (int r = 0; r < 8; ++r)
            xmid[((size_t)b * NTOK + mbase + r) * CH + ch0 + t * 16 + nb] =
                (__bf16)acc[t][r];
}

// ---------------- Stage E: out_c = X_mid @ Wp + bp --------------------------
__global__ __launch_bounds__(256) void proj_gemm(
    const __bf16* __restrict__ xmid, const float* __restrict__ Wp,
    const float* __restrict__ bp, float* __restrict__ out)
{
    __shared__ __bf16 ldsX[128 * 32];
    __shared__ __bf16 ldsW[32 * 64];
    int b = blockIdx.z;
    int n0 = blockIdx.x * 128;
    int c0 = blockIdx.y * 64;
    int tid = threadIdx.x, lane = tid & 31, wave = tid >> 5;

    v8f acc[4];
    #pragma unroll
    for (int t = 0; t < 4; ++t)
        #pragma unroll
        for (int r = 0; r < 8; ++r) acc[t][r] = 0.f;

    for (int k0 = 0; k0 < CH; k0 += 32) {
        for (int i = tid; i < 128 * 32; i += 256) {
            int r = i >> 5, kk = i & 31;
            ldsX[i] = xmid[((size_t)b * NTOK + n0 + r) * CH + k0 + kk];
        }
        for (int i = tid; i < 32 * 64; i += 256) {
            int r = i >> 6, cc = i & 63;
            ldsW[i] = (__bf16)Wp[(k0 + r) * CH + c0 + cc];
        }
        __syncthreads();

        v16bf a;
        int m = lane & 15;
        #pragma unroll
        for (int j = 0; j < 16; ++j)
            a[j] = ldsX[(wave * 16 + m) * 32 + a_kidx(lane, j)];
        #pragma unroll
        for (int t = 0; t < 4; ++t) {
            v16bf bb;
            #pragma unroll
            for (int j = 0; j < 16; ++j) bb[j] = ldsW[a_kidx(lane, j) * 64 + t * 16 + (lane & 15)];
            acc[t] = __builtin_amdgcn_wmma_f32_16x16x32_bf16(
                false, a, false, bb, (short)0, acc[t], false, false);
        }
        __syncthreads();
    }

    int mbase = n0 + wave * 16 + ((lane >= 16) ? 8 : 0);
    int nb = lane & 15;
    #pragma unroll
    for (int t = 0; t < 4; ++t) {
        float bias = bp[c0 + t * 16 + nb];
        #pragma unroll
        for (int r = 0; r < 8; ++r)
            out[((size_t)b * NTOK + mbase + r) * CH + c0 + t * 16 + nb] =
                acc[t][r] + bias;
    }
}

// ---------------- Stage F: fused dwconv3 -> GELU -> dwconv3, add into out ---
// With width-1 spatial + pad 1, only the middle kernel column contributes:
// a 1-D 3-tap conv over tokens per channel.
__global__ __launch_bounds__(256) void pos_conv_add(
    const __bf16* __restrict__ v_ws, const float* __restrict__ w1,
    const float* __restrict__ w2, float* __restrict__ out)
{
    int b  = blockIdx.y;
    int t0 = blockIdx.x * 128;
    int c  = threadIdx.x;
    float w10 = w1[c * 9 + 0 * 3 + 1], w11 = w1[c * 9 + 1 * 3 + 1], w12 = w1[c * 9 + 2 * 3 + 1];
    float w20 = w2[c * 9 + 0 * 3 + 1], w21 = w2[c * 9 + 1 * 3 + 1], w22 = w2[c * 9 + 2 * 3 + 1];

    auto loadv = [&](int n) -> float {
        if (n < 0 || n >= NTOK) return 0.f;
        return (float)v_ws[((size_t)b * NTOK + n) * CH + c];
    };
    auto gelu = [](float x) {
        return 0.5f * x * (1.f + erff(x * 0.70710678118654752440f));
    };

    float vm1 = loadv(t0 - 2);
    float v0  = loadv(t0 - 1);
    float g2 = 0.f, g1 = 0.f;
    for (int idx = 0; idx < 130; ++idx) {
        int m = t0 - 1 + idx;
        float vp1 = loadv(m + 1);
        float g0 = 0.f;
        if (m >= 0 && m < NTOK)
            g0 = gelu(w10 * vm1 + w11 * v0 + w12 * vp1);
        if (idx >= 2) {
            int tt = m - 1;                       // t0 .. t0+127
            float p = w20 * g2 + w21 * g1 + w22 * g0;
            size_t oidx = ((size_t)b * NTOK + tt) * CH + c;
            out[oidx] += p;
        }
        g2 = g1; g1 = g0; vm1 = v0; v0 = vp1;
    }
}

extern "C" void kernel_launch(void* const* d_in, const int* in_sizes, int n_in,
                              void* d_out, int out_size, void* d_ws, size_t ws_size,
                              hipStream_t stream) {
    (void)in_sizes; (void)n_in; (void)out_size; (void)ws_size;
    const float* x_in    = (const float*)d_in[0];
    const float* illu    = (const float*)d_in[1];
    const float* Wq      = (const float*)d_in[2];
    const float* Wk      = (const float*)d_in[3];
    const float* Wv      = (const float*)d_in[4];
    const float* rescale = (const float*)d_in[5];
    const float* Wp      = (const float*)d_in[6];
    const float* bp      = (const float*)d_in[7];
    const float* c1w     = (const float*)d_in[8];
    const float* c2w     = (const float*)d_in[9];
    float* out = (float*)d_out;

    char* ws = (char*)d_ws;
    size_t elems = (size_t)BATCH * NTOK * CH;       // 33.5M
    __bf16* q_ws = (__bf16*)(ws);
    __bf16* k_ws = (__bf16*)(ws + elems * 2);
    __bf16* v_ws = (__bf16*)(ws + elems * 4);
    __bf16* xmid = (__bf16*)(ws + elems * 6);
    float*  attn = (float*)(ws + elems * 8);
    float*  ssq  = attn + (size_t)BATCH * HEADS * DHEAD * DHEAD;
    float*  ssk  = ssq + BATCH * CH;

    zero_f32<<<dim3(16), dim3(256), 0, stream>>>(ssq, 2 * BATCH * CH);
    qkv_gemm<<<dim3(NTOK / 128, 12, BATCH), dim3(256), 0, stream>>>(
        x_in, Wq, Wk, Wv, q_ws, k_ws, v_ws);
    col_sumsq<<<dim3(NTOK / 64, BATCH), dim3(256), 0, stream>>>(
        q_ws, k_ws, ssq, ssk);
    make_invnorm<<<dim3((2 * BATCH * CH + 255) / 256), dim3(256), 0, stream>>>(
        ssq, 2 * BATCH * CH);
    ktq_gemm<<<dim3(BATCH * HEADS), dim3(256), 0, stream>>>(
        q_ws, k_ws, ssq, ssk, rescale, attn);
    attn_softmax<<<dim3(BATCH * HEADS * DHEAD), dim3(64), 0, stream>>>(attn);
    av_gemm<<<dim3(NTOK / 128, HEADS, BATCH), dim3(256), 0, stream>>>(
        v_ws, illu, attn, xmid);
    proj_gemm<<<dim3(NTOK / 128, CH / 64, BATCH), dim3(256), 0, stream>>>(
        xmid, Wp, bp, out);
    pos_conv_add<<<dim3(NTOK / 128, BATCH), dim3(256), 0, stream>>>(
        v_ws, c1w, c2w, out);
}
